// BoundaryTransformer_Block_11098195493244
// MI455X (gfx1250) — matrix operand order repaired and checked by reading, not compiled, not run on hardware
//
#include <hip/hip_runtime.h>
#include <hip/hip_bf16.h>

// ---------------------------------------------------------------------------
// BoundaryTransformer block for MI455X (gfx1250):
//  - all GEMMs via v_wmma_f32_16x16x32_f16 (fp16 operands, fp32 accum)
//  - GEMM B-panels staged into LDS by the Tensor Data Mover
//    (tensor_load_to_lds, double-buffered, s_wait_tensorcnt)
//  - flash attention with WMMA QK^T / PV, wave32 shuffle softmax,
//    LDS P-tile round trip guarded by s_wait_dscnt
// ---------------------------------------------------------------------------

typedef __attribute__((ext_vector_type(16))) _Float16 v16h;
typedef __attribute__((ext_vector_type(8)))  _Float16 v8h;
typedef __attribute__((ext_vector_type(8)))  float    v8f;
typedef __attribute__((ext_vector_type(4)))  unsigned u32x4;
typedef __attribute__((ext_vector_type(8)))  unsigned u32x8;

union AFrag { v16h v; v8h h[2]; };

#define BZ   4
#define SEQ  1024
#define HID  512
#define NH   8
#define HD   64
#define PROJ 2048
#define NL   2
#define MTOK (BZ * SEQ)          // 4096 token rows

__device__ __forceinline__ v8f wmma16(v16h a, v16h b, v8f c) {
  return __builtin_amdgcn_wmma_f32_16x16x32_f16(false, a, false, b,
                                                (short)0, c, false, false);
}

// LDS byte address of a __shared__ object (flat addr low 32 bits == LDS offset)
__device__ __forceinline__ unsigned ldsoff(const void* p) {
  return (unsigned)(size_t)p;
}

// Issue a TDM load of a 64-row x 32-half tile (row stride K halfs) into LDS.
// D# group0/group1 built per CDNA5 ISA 8.3/8.4; 2D tensor -> VADDR2/3 omitted.
__device__ __forceinline__ void tdm_issue_b_panel(const _Float16* gsrc,
                                                  unsigned lds_byte,
                                                  int K, int N) {
  unsigned long long ga = (unsigned long long)(size_t)gsrc;
  u32x4 g0;
  g0[0] = 1u;                                            // count = 1 (valid)
  g0[1] = lds_byte;                                      // lds_addr
  g0[2] = (unsigned)ga;                                  // global_addr[31:0]
  g0[3] = (unsigned)((ga >> 32) & 0x01FFFFFFu)           // global_addr[56:32]
          | (2u << 30);                                  // type = 2 (image)
  u32x8 g1;
  g1[0] = 1u << 16;                                      // data_size = 2 bytes
  g1[1] = ((unsigned)K & 0xFFFFu) << 16;                 // tensor_dim0 [15:0]
  g1[2] = (((unsigned)K >> 16) & 0xFFFFu)                // tensor_dim0 [31:16]
          | (((unsigned)N & 0xFFFFu) << 16);             // tensor_dim1 [15:0]
  g1[3] = (((unsigned)N >> 16) & 0xFFFFu)                // tensor_dim1 [31:16]
          | (32u << 16);                                 // tile_dim0 = 32
  g1[4] = 64u;                                           // tile_dim1 = 64
  g1[5] = (unsigned)K;                                   // dim0_stride [31:0]
  g1[6] = 0u;                                            // dim0_stride hi, d1s lo
  g1[7] = 0u;
  asm volatile("tensor_load_to_lds %0, %1" :: "s"(g0), "s"(g1) : "memory");
}

// --------------------------- weight convert+transpose ----------------------
// W (K x N, fp32) -> Wt (N x K, fp16) so WMMA B-fragments are contiguous.
__global__ void convt_kernel(const float* __restrict__ W,
                             _Float16* __restrict__ Wt, int K, int N) {
  int idx = blockIdx.x * 256 + threadIdx.x;
  if (idx >= K * N) return;
  int k = idx / N, n = idx % N;
  Wt[(size_t)n * K + k] = (_Float16)W[(size_t)k * N + n];
}

// --------------------------------- layernorm -------------------------------
__global__ __launch_bounds__(256)
void layernorm_kernel(const float* __restrict__ X, const float* __restrict__ g,
                      const float* __restrict__ bt, _Float16* __restrict__ Y,
                      int rows) {
  int wave = threadIdx.x >> 5, lane = threadIdx.x & 31;
  int row = blockIdx.x * 8 + wave;
  if (row >= rows) return;
  const float* x = X + (size_t)row * HID;
  float vals[16], s = 0.f;
#pragma unroll
  for (int j = 0; j < 16; ++j) { vals[j] = x[j * 32 + lane]; s += vals[j]; }
#pragma unroll
  for (int off = 1; off < 32; off <<= 1) s += __shfl_xor(s, off, 32);
  float mean = s * (1.f / HID);
  float vs = 0.f;
#pragma unroll
  for (int j = 0; j < 16; ++j) { float d = vals[j] - mean; vs += d * d; }
#pragma unroll
  for (int off = 1; off < 32; off <<= 1) vs += __shfl_xor(vs, off, 32);
  float inv = rsqrtf(vs * (1.f / HID) + 1e-5f);
  _Float16* y = Y + (size_t)row * HID;
#pragma unroll
  for (int j = 0; j < 16; ++j) {
    int e = j * 32 + lane;
    y[e] = (_Float16)((vals[j] - mean) * inv * g[e] + bt[e]);
  }
}

// ----------------------------------- GEMM ----------------------------------
// C[M,N] = A[M,K](f16) @ Wt[N,K](f16)^T + bias; optional ReLU; store fp16 or
// fp32 residual-accumulate. Block = 4 waves, each wave owns a 16-row M tile
// and computes a 16x64 output slab (A fragment reused across 4 N-tiles).
// The shared 64x32 B panel is DMA'd into LDS by the TDM, double-buffered.
__global__ __launch_bounds__(128)
void gemm_wmma_kernel(const _Float16* __restrict__ A,
                      const _Float16* __restrict__ Wt,
                      const float* __restrict__ bias,
                      _Float16* __restrict__ C16, float* __restrict__ Acc32,
                      int M, int K, int N, int relu) {
  __shared__ _Float16 bpanel[2][64 * 32];           // 2 x 4 KB
  const int lane = threadIdx.x & 31;
  const int wave = threadIdx.x >> 5;
  const int lo = lane & 15, hi = lane >> 4;
  const int m0 = (blockIdx.y * 4 + wave) * 16;
  const int n0 = blockIdx.x * 64;

  const _Float16* arow = A + (size_t)(m0 + lo) * K;
  const _Float16* wbase = Wt + (size_t)n0 * K;
  v8f acc[4] = {};

  if (wave == 0)
    tdm_issue_b_panel(wbase, ldsoff(&bpanel[0][0]), K, N);

  int p = 0;
  for (int k0 = 0; k0 < K; k0 += 32) {
    if (wave == 0) {
      if (k0 + 32 < K) {                            // prefetch next panel
        tdm_issue_b_panel(wbase + (k0 + 32), ldsoff(&bpanel[p ^ 1][0]), K, N);
        __builtin_amdgcn_s_wait_tensorcnt(1);       // oldest (current) done
      } else {
        __builtin_amdgcn_s_wait_tensorcnt(0);
      }
    }
    __syncthreads();                                // publish panel p
    __builtin_prefetch(arow + k0 + 256, 0, 0);      // global_prefetch (A)
    AFrag a;
    a.h[0] = *(const v8h*)(arow + k0 + hi * 16);
    a.h[1] = *(const v8h*)(arow + k0 + hi * 16 + 8);
#pragma unroll
    for (int nt = 0; nt < 4; ++nt) {
      v16h b = *(const v16h*)(&bpanel[p][(nt * 16 + lo) * 32 + hi * 16]);
      acc[nt] = wmma16(a.v, b, acc[nt]);
    }
    __syncthreads();                                // panel p consumed
    p ^= 1;
  }
#pragma unroll
  for (int nt = 0; nt < 4; ++nt) {
    int col = n0 + nt * 16 + lo;
    float bv = bias ? bias[col] : 0.f;
#pragma unroll
    for (int r = 0; r < 8; ++r) {
      float v = acc[nt][r] + bv;
      if (relu) v = fmaxf(v, 0.f);
      size_t idx = (size_t)(m0 + r + hi * 8) * N + col;
      if (Acc32) Acc32[idx] += v;                   // fp32 residual master
      else       C16[idx]   = (_Float16)v;
    }
  }
}

// -------------------------------- attention --------------------------------
// qkv16: [4096, 1536] fp16 rows; cols [0,512)=Q, [512,1024)=K, [1024,1536)=V.
#define AW 4
__global__ __launch_bounds__(128)
void attn_wmma_kernel(const _Float16* __restrict__ qkv,
                      _Float16* __restrict__ av) {
  __shared__ _Float16 plds[AW][16 * 32];
  const int lane = threadIdx.x & 31, wave = threadIdx.x >> 5;
  const int lo = lane & 15, hi = lane >> 4;
  const int b = blockIdx.z, h = blockIdx.y;
  const int qBase = (blockIdx.x * AW + wave) * 16;
  const size_t rowBase = (size_t)b * SEQ;

  const _Float16* qrow = qkv + (rowBase + qBase + lo) * (3 * HID) + h * HD;
  AFrag aq[2];
#pragma unroll
  for (int c = 0; c < 2; ++c) {
    aq[c].h[0] = *(const v8h*)(qrow + c * 32 + hi * 16);
    aq[c].h[1] = *(const v8h*)(qrow + c * 32 + hi * 16 + 8);
  }

  float m[8], l[8];
  v8f o[4] = {};
#pragma unroll
  for (int r = 0; r < 8; ++r) { m[r] = -1e30f; l[r] = 0.f; }

  for (int j0 = 0; j0 <= qBase + 15; j0 += 32) {
    v8f s[2];
#pragma unroll
    for (int nt = 0; nt < 2; ++nt) {
      int key = j0 + nt * 16 + lo;
      int keyc = key < (SEQ - 1) ? key : (SEQ - 1);
      const _Float16* krow =
          qkv + (rowBase + keyc) * (3 * HID) + HID + h * HD;
      v8f t = {};
      t = wmma16(aq[0].v, *(const v16h*)(krow + hi * 16), t);
      t = wmma16(aq[1].v, *(const v16h*)(krow + 32 + hi * 16), t);
      s[nt] = t;
    }
#pragma unroll
    for (int r = 0; r < 8; ++r) {
      int q = qBase + r + hi * 8;
      float s0 = s[0][r] * 0.125f;
      float s1 = s[1][r] * 0.125f;
      if (j0 + lo > q)      s0 = -1e30f;
      if (j0 + 16 + lo > q) s1 = -1e30f;
      float cm = fmaxf(s0, s1);
#pragma unroll
      for (int off = 1; off < 16; off <<= 1)
        cm = fmaxf(cm, __shfl_xor(cm, off, 32));
      float mn = fmaxf(m[r], cm);
      float p0 = __expf(s0 - mn), p1 = __expf(s1 - mn);
      float ps = p0 + p1;
#pragma unroll
      for (int off = 1; off < 16; off <<= 1) ps += __shfl_xor(ps, off, 32);
      float sc = __expf(m[r] - mn);
      l[r] = l[r] * sc + ps;
      m[r] = mn;
      o[0][r] *= sc; o[1][r] *= sc; o[2][r] *= sc; o[3][r] *= sc;
      plds[wave][(r + hi * 8) * 32 + lo]      = (_Float16)p0;
      plds[wave][(r + hi * 8) * 32 + 16 + lo] = (_Float16)p1;
    }
    asm volatile("s_wait_dscnt 0" ::: "memory");    // cross-lane P visibility
    AFrag ap;
    ap.h[0] = *(const v8h*)(&plds[wave][lo * 32 + hi * 8]);
    ap.h[1] = *(const v8h*)(&plds[wave][lo * 32 + 16 + hi * 8]);
#pragma unroll
    for (int dt = 0; dt < 4; ++dt) {
      v16h bv;
#pragma unroll
      for (int i = 0; i < 16; ++i) {
        int key = j0 + (i < 8 ? hi * 8 + i : 16 + hi * 8 + (i - 8));
        int keyc = key < (SEQ - 1) ? key : (SEQ - 1);
        bv[i] = qkv[(rowBase + keyc) * (3 * HID) + 2 * HID + h * HD +
                    dt * 16 + lo];
      }
      o[dt] = wmma16(ap.v, bv, o[dt]);
    }
  }
#pragma unroll
  for (int dt = 0; dt < 4; ++dt)
#pragma unroll
    for (int r = 0; r < 8; ++r) {
      size_t idx = (rowBase + qBase + r + hi * 8) * HID + h * HD + dt * 16 + lo;
      av[idx] = (_Float16)(o[dt][r] / l[r]);
    }
}

// ------------------------- boundary decision + blend -----------------------
__global__ __launch_bounds__(256)
void boundary_kernel(const float* __restrict__ out, const float* __restrict__ lx,
                     const float* __restrict__ mm, const float* __restrict__ bdw,
                     const float* __restrict__ bdb, float* __restrict__ newout,
                     float* __restrict__ newmask, int rows) {
  int wave = threadIdx.x >> 5, lane = threadIdx.x & 31;
  int row = blockIdx.x * 8 + wave;
  if (row >= rows) return;
  const float* x = out + (size_t)row * HID;
  float dot = 0.f;
#pragma unroll
  for (int j = 0; j < 16; ++j) dot += x[j * 32 + lane] * bdw[j * 32 + lane];
#pragma unroll
  for (int off = 1; off < 32; off <<= 1) dot += __shfl_xor(dot, off, 32);
  dot += bdb[0];
  float mk = dot > 0.f ? 1.f : 0.f;       // sigmoid(x) > 0.5  <=>  x > 0
  float mmv = mm[row];
  const float* l0 = lx + (size_t)row * HID;
  float* y = newout + (size_t)row * HID;
#pragma unroll
  for (int j = 0; j < 16; ++j) {
    int e = j * 32 + lane;
    y[e] = x[e] * mmv + (1.f - mmv) * l0[e];
  }
  if (lane == 0) newmask[row] = mk;
}

// --------------------------------- driver ----------------------------------
static inline void launch_gemm(const _Float16* A, const _Float16* Wt,
                               const float* bias, _Float16* C16, float* Acc32,
                               int M, int K, int N, int relu, hipStream_t s) {
  dim3 g(N / 64, M / 64);
  gemm_wmma_kernel<<<g, 128, 0, s>>>(A, Wt, bias, C16, Acc32, M, K, N, relu);
}

extern "C" void kernel_launch(void* const* d_in, const int* in_sizes, int n_in,
                              void* d_out, int out_size, void* d_ws,
                              size_t ws_size, hipStream_t stream) {
  const float* layer_x = (const float*)d_in[0];
  // d_in[1] full_mask: causal, reimplemented analytically
  const float* mask_mult = (const float*)d_in[2];
  const float* qkv_w = (const float*)d_in[3];
  const float* qkv_b = (const float*)d_in[4];
  const float* o_w   = (const float*)d_in[5];
  const float* o_b   = (const float*)d_in[6];
  const float* ln1_g = (const float*)d_in[7];
  const float* ln1_b = (const float*)d_in[8];
  const float* ff_w1 = (const float*)d_in[9];
  const float* ff_b1 = (const float*)d_in[10];
  const float* ff_w2 = (const float*)d_in[11];
  const float* ff_b2 = (const float*)d_in[12];
  const float* ln2_g = (const float*)d_in[13];
  const float* ln2_b = (const float*)d_in[14];
  const float* bd_w  = (const float*)d_in[15];
  const float* bd_b  = (const float*)d_in[16];

  char* p = (char*)d_ws;
  auto carve = [&](size_t bytes) {
    void* q = (void*)p;
    p += (bytes + 255) & ~(size_t)255;
    return q;
  };
  float*     out32  = (float*)    carve((size_t)MTOK * HID * 4);
  _Float16*  h16    = (_Float16*) carve((size_t)MTOK * HID * 2);
  _Float16*  qkv16  = (_Float16*) carve((size_t)MTOK * 3 * HID * 2);
  _Float16*  av16   = (_Float16*) carve((size_t)MTOK * HID * 2);
  _Float16*  ff16   = (_Float16*) carve((size_t)MTOK * PROJ * 2);
  _Float16*  qkvWt  = (_Float16*) carve((size_t)NL * 3 * HID * HID * 2);
  _Float16*  oWt    = (_Float16*) carve((size_t)NL * HID * HID * 2);
  _Float16*  ff1Wt  = (_Float16*) carve((size_t)NL * HID * PROJ * 2);
  _Float16*  ff2Wt  = (_Float16*) carve((size_t)NL * PROJ * HID * 2);
  (void)ws_size; (void)n_in; (void)in_sizes; (void)out_size;

  auto convt = [&](const float* W, _Float16* Wt, int K, int N) {
    int n = K * N;
    convt_kernel<<<(n + 255) / 256, 256, 0, stream>>>(W, Wt, K, N);
  };
  for (int i = 0; i < NL; ++i) {
    convt(qkv_w + (size_t)i * HID * 3 * HID, qkvWt + (size_t)i * 3 * HID * HID,
          HID, 3 * HID);
    convt(o_w + (size_t)i * HID * HID, oWt + (size_t)i * HID * HID, HID, HID);
    convt(ff_w1 + (size_t)i * HID * PROJ, ff1Wt + (size_t)i * PROJ * HID,
          HID, PROJ);
    convt(ff_w2 + (size_t)i * PROJ * HID, ff2Wt + (size_t)i * HID * PROJ,
          PROJ, HID);
  }

  hipMemcpyAsync(out32, layer_x, (size_t)MTOK * HID * 4,
                 hipMemcpyDeviceToDevice, stream);

  for (int i = 0; i < NL; ++i) {
    layernorm_kernel<<<MTOK / 8, 256, 0, stream>>>(
        out32, ln1_g + i * HID, ln1_b + i * HID, h16, MTOK);
    launch_gemm(h16, qkvWt + (size_t)i * 3 * HID * HID, qkv_b + i * 3 * HID,
                qkv16, nullptr, MTOK, HID, 3 * HID, 0, stream);
    attn_wmma_kernel<<<dim3(SEQ / (16 * AW), NH, BZ), 128, 0, stream>>>(
        qkv16, av16);
    launch_gemm(av16, oWt + (size_t)i * HID * HID, o_b + i * HID, nullptr,
                out32, MTOK, HID, HID, 0, stream);
    layernorm_kernel<<<MTOK / 8, 256, 0, stream>>>(
        out32, ln2_g + i * HID, ln2_b + i * HID, h16, MTOK);
    launch_gemm(h16, ff1Wt + (size_t)i * PROJ * HID, ff_b1 + i * PROJ, ff16,
                nullptr, MTOK, HID, PROJ, 1, stream);
    launch_gemm(ff16, ff2Wt + (size_t)i * HID * PROJ, ff_b2 + i * HID, nullptr,
                out32, MTOK, PROJ, HID, 0, stream);
  }

  float* new_out  = (float*)d_out;
  float* new_mask = new_out + (size_t)MTOK * HID;
  boundary_kernel<<<MTOK / 8, 256, 0, stream>>>(
      out32, layer_x, mask_mult, bd_w, bd_b, new_out, new_mask, MTOK);
}
